// AlignmentModule_70566312673501
// MI455X (gfx1250) — matrix operand order repaired and checked by reading, not compiled, not run on hardware
//
#include <hip/hip_runtime.h>

// ---------------------------------------------------------------------------
// Problem constants
// ---------------------------------------------------------------------------
constexpr int B_      = 8;
constexpr int TT      = 1024;   // T_TEXT
constexpr int TF      = 4096;   // T_FEATS
constexpr int AD      = 256;    // ADIM
constexpr int OD      = 80;     // ODIM
constexpr int ODP     = 96;     // ODIM padded to multiple of 32 for K-tiling
constexpr float TEMP  = 0.0005f;
constexpr float EPS_  = 1e-8f;

typedef __attribute__((ext_vector_type(16))) __bf16 v16bf;
typedef __attribute__((ext_vector_type(8)))  __bf16 v8bf;
typedef __attribute__((ext_vector_type(8)))  float  v8f;
typedef __attribute__((ext_vector_type(4)))  unsigned int u32x4;
typedef __attribute__((ext_vector_type(8)))  int i32x8;
typedef __attribute__((ext_vector_type(4)))  int i32x4;

// Tensor Data Mover availability (arity differs across toolchains)
#if defined(__HIP_DEVICE_COMPILE__) && __has_builtin(__builtin_amdgcn_tensor_load_to_lds)
#define USE_TDM 1
#else
#define USE_TDM 0
#endif

// ---------------------------------------------------------------------------
// Helpers
// ---------------------------------------------------------------------------
__device__ __forceinline__ __bf16 f2bf(float f) {
    unsigned u = __float_as_uint(f);
    u += 0x7fffu + ((u >> 16) & 1u);            // round-to-nearest-even
    unsigned short h = (unsigned short)(u >> 16);
    return __builtin_bit_cast(__bf16, h);
}

// A-fragment (16(M) x 32(K) bf16), source row-major, row stride ld elements.
// Lane L: M = row0 + (L&15).  elems 0..7 -> K = k0 + (L>=16?8:0) + 0..7
//                             elems 8..15 -> K = k0 + 16 + (L>=16?8:0) + 0..7
template <typename P>
__device__ __forceinline__ v16bf load_frag_A(P src, int ld, int row0, int k0, int lane) {
    const __bf16* p = &src[(size_t)(row0 + (lane & 15)) * ld + k0 + ((lane >> 4) << 3)];
    v8bf lo = *(const v8bf*)p;
    v8bf hi = *(const v8bf*)(p + 16);
    v16bf r;
#pragma unroll
    for (int i = 0; i < 8; ++i) { r[i] = lo[i]; r[i + 8] = hi[i]; }
    return r;
}

// B-fragment (32(K) x 16(N) bf16) from N-major storage (row = N, K contiguous).
// Lane L: N = row0 + (L&15).  elems 0..7 -> K = k0 + (L>=16?8:0) + 0..7
//                             elems 8..15 -> K = k0 + 16 + (L>=16?8:0) + 0..7
template <typename P>
__device__ __forceinline__ v16bf load_frag_B(P src, int ld, int row0, int k0, int lane) {
    const __bf16* p = &src[(size_t)(row0 + (lane & 15)) * ld + k0 + ((lane >> 4) << 3)];
    v8bf lo = *(const v8bf*)p;
    v8bf hi = *(const v8bf*)(p + 16);
    v16bf r;
#pragma unroll
    for (int i = 0; i < 8; ++i) { r[i] = lo[i]; r[i + 8] = hi[i]; }
    return r;
}

__device__ __forceinline__ v8f wmma_bf16(v16bf a, v16bf b, v8f c) {
    return __builtin_amdgcn_wmma_f32_16x16x32_bf16(false, a, false, b, (short)0, c,
                                                   false, false);
}

// ---------------------------------------------------------------------------
// Tiny prep kernels
// ---------------------------------------------------------------------------
__global__ void spk_proj_kernel(const float* __restrict__ se,
                                const float* __restrict__ tw,   // (AD, AD)
                                const float* __restrict__ fw,   // (OD, AD)
                                float* __restrict__ spk_text,   // (B, AD)
                                float* __restrict__ spk_feat) { // (B, OD)
    int b = blockIdx.x, c = threadIdx.x;
    float s = 0.f;
    for (int a = 0; a < AD; ++a) s += se[b * AD + a] * tw[c * AD + a];
    spk_text[b * AD + c] = s;
    if (c < OD) {
        float s2 = 0.f;
        for (int a = 0; a < AD; ++a) s2 += se[b * AD + a] * fw[c * AD + a];
        spk_feat[b * OD + c] = s2;
    }
}

// dst[cout][k*cip + ci] = (ci<cin) ? w[(cout*cin + ci)*3 + k] : 0   (bf16)
__global__ void pack_w3_kernel(const float* __restrict__ w, __bf16* __restrict__ dst,
                               int cin, int cip) {
    int total = 256 * 3 * cip;
    for (int i = blockIdx.x * blockDim.x + threadIdx.x; i < total;
         i += gridDim.x * blockDim.x) {
        int cout = i / (3 * cip);
        int rem  = i % (3 * cip);
        int k = rem / cip, ci = rem % cip;
        float v = (ci < cin) ? w[((size_t)cout * cin + ci) * 3 + k] : 0.f;
        dst[(size_t)cout * (3 * cip) + k * cip + ci] = f2bf(v);
    }
}

__global__ void pack_w1x1_kernel(const float* __restrict__ w, __bf16* __restrict__ dst) {
    int i = blockIdx.x * blockDim.x + threadIdx.x;
    if (i < 256 * 256) dst[i] = f2bf(w[i]);
}

// ---------------------------------------------------------------------------
// Text encoder: conv3 -> relu -> conv1 ; outputs bf16 enc (B,TT,AD) and t2
// Block: 256 threads (8 waves), tile of 64 text positions.
// ---------------------------------------------------------------------------
__global__ __launch_bounds__(256, 1) void text_enc_kernel(
        const float* __restrict__ texts, const float* __restrict__ spk_text,
        const __bf16* __restrict__ w1p, const float* __restrict__ b1,
        const __bf16* __restrict__ w2p, const float* __restrict__ b2,
        __bf16* __restrict__ enc_out, float* __restrict__ t2_out) {
    __shared__ __bf16 Xc[66][AD];     // rows: position t0-1+s
    __shared__ __bf16 H1[64][AD];
    __shared__ float  sq[64];

    const int b = blockIdx.x / (TT / 64);
    const int t0 = (blockIdx.x % (TT / 64)) * 64;
    const int tid = threadIdx.x;
    const int lane = tid & 31, w = tid >> 5;

    for (int i = tid; i < 66 * AD; i += 256) {
        int s = i >> 8, ci = i & 255;
        int pos = t0 - 1 + s;
        float v = 0.f;
        if (pos >= 0 && pos < TT)
            v = texts[((size_t)b * TT + pos) * AD + ci] + spk_text[b * AD + ci];
        Xc[s][ci] = f2bf(v);
    }
    if (tid < 64) sq[tid] = 0.f;
    __syncthreads();

    // GEMM1: H1 = relu(im2col(Xc) * W1^T + b1),  M=64 N=256 K=768
#pragma unroll
    for (int i = 0; i < 8; ++i) {
        int idx = w * 8 + i, mt = idx >> 4, nt = idx & 15;
        float bias = b1[nt * 16 + (lane & 15)];
        v8f acc;
#pragma unroll
        for (int v = 0; v < 8; ++v) acc[v] = bias;
        for (int kk = 0; kk < 768; kk += 32) {
            int ktap = kk >> 8, ci0 = kk & 255;
            v16bf a  = load_frag_A(&Xc[mt * 16 + ktap][0], AD, 0, ci0, lane);
            v16bf bb = load_frag_B(w1p, 768, nt * 16, kk, lane);
            acc = wmma_bf16(a, bb, acc);
        }
#pragma unroll
        for (int v = 0; v < 8; ++v) {
            int m = mt * 16 + v + ((lane >> 4) << 3);
            float x = acc[v];
            H1[m][nt * 16 + (lane & 15)] = f2bf(x > 0.f ? x : 0.f);
        }
    }
    __syncthreads();

    // GEMM2: enc = H1 * W2^T + b2,  K=256 ; also t2 = sum_c enc^2
#pragma unroll
    for (int i = 0; i < 8; ++i) {
        int idx = w * 8 + i, mt = idx >> 4, nt = idx & 15;
        float bias = b2[nt * 16 + (lane & 15)];
        v8f acc;
#pragma unroll
        for (int v = 0; v < 8; ++v) acc[v] = bias;
        for (int kk = 0; kk < 256; kk += 32) {
            v16bf a  = load_frag_A(&H1[mt * 16][0], AD, 0, kk, lane);
            v16bf bb = load_frag_B(w2p, 256, nt * 16, kk, lane);
            acc = wmma_bf16(a, bb, acc);
        }
#pragma unroll
        for (int v = 0; v < 8; ++v) {
            int m = mt * 16 + v + ((lane >> 4) << 3);
            float e = acc[v];
            enc_out[((size_t)b * TT + t0 + m) * AD + nt * 16 + (lane & 15)] = f2bf(e);
            float s2 = e * e;
            s2 += __shfl_xor(s2, 1, 32); s2 += __shfl_xor(s2, 2, 32);
            s2 += __shfl_xor(s2, 4, 32); s2 += __shfl_xor(s2, 8, 32);
            if ((lane & 15) == 0) atomicAdd(&sq[m], s2);
        }
    }
    __syncthreads();
    if (tid < 64) t2_out[b * TT + t0 + tid] = sq[tid];
}

// ---------------------------------------------------------------------------
// Feat encoder: conv3 -> relu -> conv3 -> relu -> conv1 ; outputs enc + f2
// ---------------------------------------------------------------------------
__global__ __launch_bounds__(256, 1) void feat_enc_kernel(
        const float* __restrict__ feats, const float* __restrict__ spk_feat,
        const __bf16* __restrict__ wf1p, const float* __restrict__ fb1,
        const __bf16* __restrict__ wf2p, const float* __restrict__ fb2,
        const __bf16* __restrict__ wf3p, const float* __restrict__ fb3,
        __bf16* __restrict__ enc_out, float* __restrict__ f2_out) {
    __shared__ __bf16 Xf[80][ODP];    // rows: position t0-2+s (s<68 valid)
    __shared__ __bf16 H1[80][AD];     // rows: position t0-1+r (r<66 valid)
    __shared__ __bf16 H2[64][AD];
    __shared__ float  sq[64];

    const int b = blockIdx.x / (TF / 64);
    const int t0 = (blockIdx.x % (TF / 64)) * 64;
    const int tid = threadIdx.x;
    const int lane = tid & 31, w = tid >> 5;

    for (int i = tid; i < 80 * ODP; i += 256) {
        int s = i / ODP, ci = i % ODP;
        int pos = t0 - 2 + s;
        float v = 0.f;
        if (s < 68 && ci < OD && pos >= 0 && pos < TF)
            v = feats[((size_t)b * TF + pos) * OD + ci] + spk_feat[b * OD + ci];
        Xf[s][ci] = f2bf(v);
    }
    if (tid < 64) sq[tid] = 0.f;
    __syncthreads();

    // conv1: M=80 (66 useful), N=256, K=288
#pragma unroll
    for (int i = 0; i < 10; ++i) {
        int idx = w * 10 + i, mt = idx >> 4, nt = idx & 15;
        float bias = fb1[nt * 16 + (lane & 15)];
        v8f acc;
#pragma unroll
        for (int v = 0; v < 8; ++v) acc[v] = bias;
        for (int kk = 0; kk < 288; kk += 32) {
            int ktap = kk / ODP, ci0 = kk % ODP;
            v16bf a  = load_frag_A(&Xf[mt * 16 + ktap][0], ODP, 0, ci0, lane);
            v16bf bb = load_frag_B(wf1p, 288, nt * 16, kk, lane);
            acc = wmma_bf16(a, bb, acc);
        }
#pragma unroll
        for (int v = 0; v < 8; ++v) {
            int m = mt * 16 + v + ((lane >> 4) << 3);
            float x = acc[v];
            if (m < 66) H1[m][nt * 16 + (lane & 15)] = f2bf(x > 0.f ? x : 0.f);
        }
    }
    __syncthreads();

    // conv2: M=64, N=256, K=768
#pragma unroll
    for (int i = 0; i < 8; ++i) {
        int idx = w * 8 + i, mt = idx >> 4, nt = idx & 15;
        float bias = fb2[nt * 16 + (lane & 15)];
        v8f acc;
#pragma unroll
        for (int v = 0; v < 8; ++v) acc[v] = bias;
        for (int kk = 0; kk < 768; kk += 32) {
            int ktap = kk >> 8, ci0 = kk & 255;
            v16bf a  = load_frag_A(&H1[mt * 16 + ktap][0], AD, 0, ci0, lane);
            v16bf bb = load_frag_B(wf2p, 768, nt * 16, kk, lane);
            acc = wmma_bf16(a, bb, acc);
        }
#pragma unroll
        for (int v = 0; v < 8; ++v) {
            int m = mt * 16 + v + ((lane >> 4) << 3);
            float x = acc[v];
            H2[m][nt * 16 + (lane & 15)] = f2bf(x > 0.f ? x : 0.f);
        }
    }
    __syncthreads();

    // conv3 (1x1): K=256 -> enc + f2
#pragma unroll
    for (int i = 0; i < 8; ++i) {
        int idx = w * 8 + i, mt = idx >> 4, nt = idx & 15;
        float bias = fb3[nt * 16 + (lane & 15)];
        v8f acc;
#pragma unroll
        for (int v = 0; v < 8; ++v) acc[v] = bias;
        for (int kk = 0; kk < 256; kk += 32) {
            v16bf a  = load_frag_A(&H2[mt * 16][0], AD, 0, kk, lane);
            v16bf bb = load_frag_B(wf3p, 256, nt * 16, kk, lane);
            acc = wmma_bf16(a, bb, acc);
        }
#pragma unroll
        for (int v = 0; v < 8; ++v) {
            int m = mt * 16 + v + ((lane >> 4) << 3);
            float e = acc[v];
            enc_out[((size_t)b * TF + t0 + m) * AD + nt * 16 + (lane & 15)] = f2bf(e);
            float s2 = e * e;
            s2 += __shfl_xor(s2, 1, 32); s2 += __shfl_xor(s2, 2, 32);
            s2 += __shfl_xor(s2, 4, 32); s2 += __shfl_xor(s2, 8, 32);
            if ((lane & 15) == 0) atomicAdd(&sq[m], s2);
        }
    }
    __syncthreads();
    if (tid < 64) f2_out[b * TF + t0 + tid] = sq[tid];
}

// ---------------------------------------------------------------------------
// Cross GEMM + fused double softmax.
// Block: 256 threads (8 waves); owns 32 feat rows x 1024 text cols.
// Wave w covers columns [w*128, w*128+128).
// The 32x1024 f32 prior tile is DMA'd into LDS by the Tensor Data Mover,
// overlapped with the whole K-loop, and consumed in the epilogue.
// ---------------------------------------------------------------------------
__global__ __launch_bounds__(256, 1) void cross_softmax_kernel(
        const __bf16* __restrict__ fenc, const __bf16* __restrict__ tenc,
        const float* __restrict__ f2g, const float* __restrict__ t2g,
        const float* __restrict__ prior, const unsigned char* __restrict__ xmask,
        float* __restrict__ out_attn, float* __restrict__ out_logp) {
#if USE_TDM
    __shared__ float priorS[32][TT];          // 128 KB TDM-filled tile
#endif
    __shared__ float t2S[TT];
    __shared__ float f2S[32];
    __shared__ unsigned char maskS[TT];
    __shared__ float red[32][8];
    __shared__ float rowA[32];   // row max / max2
    __shared__ float rowB[32];   // logZ / 1/sum2

    const int b  = blockIdx.x / (TF / 32);
    const int f0 = (blockIdx.x % (TF / 32)) * 32;
    const int tid = threadIdx.x;
    const int lane = tid & 31, w = tid >> 5;
    const int nlo = lane & 15, khi = lane >> 4;

    const float* priorBase = prior + ((size_t)b * TF + f0) * TT;

#if USE_TDM
    if (w == 0) {
        // Tensor DMA descriptor (D#), ISA ch.8: 2-D tile 1024(x) x 32(y) f32.
        unsigned long long ga = (unsigned long long)(uintptr_t)priorBase;
        unsigned int lds_addr = (unsigned int)(uintptr_t)&priorS[0][0]; // low 32 = LDS offset
        u32x4 g0;
        g0[0] = 1u;                                  // count=1, user mode
        g0[1] = lds_addr;                            // lds_addr
        g0[2] = (unsigned int)ga;                    // global_addr[31:0]
        g0[3] = (unsigned int)((ga >> 32) & 0x1FFFFFFu) | (2u << 30); // addr[56:32], type=2
        i32x8 g1;
        g1[0] = (int)(2u << 16);                     // workgroup_mask=0, data_size=2 (4B)
        g1[1] = (int)(1024u << 16);                  // tensor_dim0[15:0]=1024 in bits[63:48]
        g1[2] = (int)(32u << 16);                    // tensor_dim0 hi=0; tensor_dim1 lo=32
        g1[3] = (int)(1024u << 16);                  // tensor_dim1 hi=0; tile_dim0=1024
        g1[4] = 32;                                  // tile_dim1=32; tile_dim2=0
        g1[5] = 1024;                                // tensor_dim0_stride lo
        g1[6] = 0;                                   // stride0 hi; tensor_dim1_stride lo
        g1[7] = 0;
        i32x4 gz = {0, 0, 0, 0};
#if __clang_major__ >= 23
        i32x8 gz8 = {0, 0, 0, 0, 0, 0, 0, 0};
        __builtin_amdgcn_tensor_load_to_lds(g0, g1, gz, gz, gz8, 0);
#else
        __builtin_amdgcn_tensor_load_to_lds(g0, g1, gz, gz, 0);
#endif
    }
#else
    // fallback: prefetch the prior tile -> global_prefetch_b8
    for (int i = tid; i < 32 * 16; i += 256)
        __builtin_prefetch(priorBase + (size_t)(i >> 4) * TT + (i & 15) * 64, 0, 0);
#endif

    for (int i = tid; i < TT; i += 256) {
        t2S[i]   = t2g[b * TT + i];
        maskS[i] = xmask[(size_t)b * TT + i];
    }
    if (tid < 32) f2S[tid] = f2g[b * TF + f0 + tid];
    __syncthreads();

    const __bf16* Abase = fenc + ((size_t)b * TF + f0) * AD;
    const __bf16* Bbase = tenc + ((size_t)b * TT + w * 128) * AD;

    v8f acc[2][8];
#pragma unroll
    for (int mt = 0; mt < 2; ++mt)
#pragma unroll
        for (int nt = 0; nt < 8; ++nt)
#pragma unroll
            for (int v = 0; v < 8; ++v) acc[mt][nt][v] = 0.f;

    for (int kk = 0; kk < AD; kk += 32) {
        v16bf a0 = load_frag_A(Abase, AD, 0,  kk, lane);
        v16bf a1 = load_frag_A(Abase, AD, 16, kk, lane);
#pragma unroll
        for (int nt = 0; nt < 8; ++nt) {
            v16bf bb = load_frag_B(Bbase, AD, nt * 16, kk, lane);
            acc[0][nt] = wmma_bf16(a0, bb, acc[0][nt]);
            acc[1][nt] = wmma_bf16(a1, bb, acc[1][nt]);
        }
    }

    // logits = -TEMP * (||f||^2 + ||t||^2 - 2*cross)
#pragma unroll
    for (int mt = 0; mt < 2; ++mt)
#pragma unroll
        for (int nt = 0; nt < 8; ++nt)
#pragma unroll
            for (int v = 0; v < 8; ++v) {
                int m = mt * 16 + v + (khi << 3);
                int t = w * 128 + nt * 16 + nlo;
                acc[mt][nt][v] =
                    -TEMP * (f2S[m] + t2S[t] - 2.f * acc[mt][nt][v]);
            }

    // cross-wave row reduction helper (rows of 1024 spread over 8 waves)
    auto reduce_rows = [&](auto getval, bool isMax, float* dst) {
#pragma unroll
        for (int mt = 0; mt < 2; ++mt)
#pragma unroll
            for (int v = 0; v < 8; ++v) {
                float p = isMax ? -__builtin_inff() : 0.f;
#pragma unroll
                for (int nt = 0; nt < 8; ++nt) {
                    float x = getval(mt, nt, v);
                    p = isMax ? fmaxf(p, x) : (p + x);
                }
#pragma unroll
                for (int mk = 1; mk < 16; mk <<= 1) {
                    float o = __shfl_xor(p, mk, 32);
                    p = isMax ? fmaxf(p, o) : (p + o);
                }
                if ((lane & 15) == 0) red[mt * 16 + v + (khi << 3)][w] = p;
            }
        __syncthreads();
        if (tid < 32) {
            float r = isMax ? -__builtin_inff() : 0.f;
#pragma unroll
            for (int j = 0; j < 8; ++j)
                r = isMax ? fmaxf(r, red[tid][j]) : (r + red[tid][j]);
            dst[tid] = r;
        }
        __syncthreads();
    };

    // -------- log_softmax --------
    reduce_rows([&](int mt, int nt, int v) { return acc[mt][nt][v]; }, true, rowA);
    reduce_rows([&](int mt, int nt, int v) {
        int m = mt * 16 + v + (khi << 3);
        return expf(acc[mt][nt][v] - rowA[m]);
    }, false, rowB);
    if (tid < 32) rowB[tid] = rowA[tid] + logf(rowB[tid]);   // logZ
#if USE_TDM
    if (w == 0) __builtin_amdgcn_s_wait_tensorcnt(0);        // prior tile landed
#endif
    __syncthreads();

    float* lpBase = out_logp + ((size_t)b * TF + f0) * TT;
    float* atBase = out_attn + ((size_t)b * TF + f0) * TT;

    // attn_logprob = logits - logZ + log(prior + eps); keep in acc
#pragma unroll
    for (int mt = 0; mt < 2; ++mt)
#pragma unroll
        for (int nt = 0; nt < 8; ++nt)
#pragma unroll
            for (int v = 0; v < 8; ++v) {
                int m = mt * 16 + v + (khi << 3);
                int t = w * 128 + nt * 16 + nlo;
#if USE_TDM
                float pv = priorS[m][t];
#else
                float pv = priorBase[(size_t)m * TT + t];
#endif
                float alp = acc[mt][nt][v] - rowB[m] + logf(pv + EPS_);
                acc[mt][nt][v] = alp;
                lpBase[(size_t)m * TT + t] = alp;
            }
    __syncthreads();

    // -------- masked softmax --------
    reduce_rows([&](int mt, int nt, int v) {
        int t = w * 128 + nt * 16 + nlo;
        return maskS[t] ? -__builtin_inff() : acc[mt][nt][v];
    }, true, rowA);
    reduce_rows([&](int mt, int nt, int v) {
        int m = mt * 16 + v + (khi << 3);
        int t = w * 128 + nt * 16 + nlo;
        return maskS[t] ? 0.f : expf(acc[mt][nt][v] - rowA[m]);
    }, false, rowB);
    if (tid < 32) rowB[tid] = 1.f / rowB[tid];
    __syncthreads();

#pragma unroll
    for (int mt = 0; mt < 2; ++mt)
#pragma unroll
        for (int nt = 0; nt < 8; ++nt)
#pragma unroll
            for (int v = 0; v < 8; ++v) {
                int m = mt * 16 + v + (khi << 3);
                int t = w * 128 + nt * 16 + nlo;
                float a = maskS[t] ? 0.f
                                   : expf(acc[mt][nt][v] - rowA[m]) * rowB[m];
                atBase[(size_t)m * TT + t] = a;
            }
}

// ---------------------------------------------------------------------------
// Launch
// ---------------------------------------------------------------------------
extern "C" void kernel_launch(void* const* d_in, const int* in_sizes, int n_in,
                              void* d_out, int out_size, void* d_ws, size_t ws_size,
                              hipStream_t stream) {
    const float* texts      = (const float*)d_in[0];
    const float* feats      = (const float*)d_in[1];
    const float* prior      = (const float*)d_in[2];
    const float* se         = (const float*)d_in[3];
    const float* text_w1    = (const float*)d_in[4];
    const float* text_b1    = (const float*)d_in[5];
    const float* text_w2    = (const float*)d_in[6];
    const float* text_b2    = (const float*)d_in[7];
    const float* feat_w1    = (const float*)d_in[8];
    const float* feat_b1    = (const float*)d_in[9];
    const float* feat_w2    = (const float*)d_in[10];
    const float* feat_b2    = (const float*)d_in[11];
    const float* feat_w3    = (const float*)d_in[12];
    const float* feat_b3    = (const float*)d_in[13];
    const float* text_spk_w = (const float*)d_in[14];
    const float* feat_spk_w = (const float*)d_in[15];
    const unsigned char* xm = (const unsigned char*)d_in[16];  // jnp.bool_ = 1 byte

    char* p = (char*)d_ws;
    auto carve = [&](size_t n) {
        char* r = p; p += (n + 255) & ~(size_t)255; return r;
    };
    __bf16* tenc     = (__bf16*)carve((size_t)B_ * TT * AD * 2);
    __bf16* fenc     = (__bf16*)carve((size_t)B_ * TF * AD * 2);
    float*  t2       = (float*) carve((size_t)B_ * TT * 4);
    float*  f2       = (float*) carve((size_t)B_ * TF * 4);
    float*  spk_text = (float*) carve((size_t)B_ * AD * 4);
    float*  spk_feat = (float*) carve((size_t)B_ * OD * 4);
    __bf16* w1p      = (__bf16*)carve((size_t)256 * 768 * 2);
    __bf16* w2p      = (__bf16*)carve((size_t)256 * 256 * 2);
    __bf16* wf1p     = (__bf16*)carve((size_t)256 * 288 * 2);
    __bf16* wf2p     = (__bf16*)carve((size_t)256 * 768 * 2);
    __bf16* wf3p     = (__bf16*)carve((size_t)256 * 256 * 2);

    float* out_attn = (float*)d_out;
    float* out_logp = out_attn + (size_t)B_ * TF * TT;

    spk_proj_kernel<<<B_, 256, 0, stream>>>(se, text_spk_w, feat_spk_w,
                                            spk_text, spk_feat);
    pack_w3_kernel<<<192, 256, 0, stream>>>(text_w1, w1p, 256, 256);
    pack_w1x1_kernel<<<256, 256, 0, stream>>>(text_w2, w2p);
    pack_w3_kernel<<<96, 256, 0, stream>>>(feat_w1, wf1p, OD, ODP);
    pack_w3_kernel<<<192, 256, 0, stream>>>(feat_w2, wf2p, 256, 256);
    pack_w1x1_kernel<<<256, 256, 0, stream>>>(feat_w3, wf3p);

    text_enc_kernel<<<B_ * (TT / 64), 256, 0, stream>>>(
        texts, spk_text, w1p, text_b1, w2p, text_b2, tenc, t2);
    feat_enc_kernel<<<B_ * (TF / 64), 256, 0, stream>>>(
        feats, spk_feat, wf1p, feat_b1, wf2p, feat_b2, wf3p, feat_b3, fenc, f2);
    cross_softmax_kernel<<<B_ * (TF / 32), 256, 0, stream>>>(
        fenc, tenc, f2, t2, prior, xm, out_attn, out_logp);
}
